// MixerBlock_26113401160188
// MI455X (gfx1250) — matrix operand order repaired and checked by reading, not compiled, not run on hardware
//
#include <hip/hip_runtime.h>
#include <hip/hip_bf16.h>
#include <math.h>

// ---------------- problem constants ----------------
constexpr int kB   = 2;
constexpr int kL   = 4096;
constexpr int kD   = 1024;
constexpr int kDIN = 2048;           // E*D
constexpr int kP   = 64;
constexpr int kH   = 32;             // DIN/P
constexpr int kN   = 16;
constexpr int kKc  = 4;              // conv taps
constexpr int kDPROJ = 2*kDIN + 2*kN + kH;   // 4160
constexpr int kCONVD = kDIN + 2*kN;          // 2080
constexpr int kRows  = kB * kL;              // 8192

// ---------------- vector types ----------------
typedef __attribute__((ext_vector_type(16))) __bf16 v16bf;
typedef __attribute__((ext_vector_type(8)))  float  v8f;
typedef __attribute__((ext_vector_type(4)))  unsigned int u32x4;

union FragBf { u32x4 q[2]; v16bf v; };

// =====================================================================
// f32 -> bf16 conversion (weights)
// =====================================================================
__global__ void cvt_bf16_kernel(const float* __restrict__ in, __bf16* __restrict__ out, int n) {
  int i = blockIdx.x * blockDim.x + threadIdx.x;
  if (i < n) out[i] = (__bf16)in[i];
}

// =====================================================================
// RMSNorm row kernel: out_bf16[row, :] = x * rsqrt(mean(x^2)+eps) * w
// =====================================================================
__global__ __launch_bounds__(256) void rms_bf16_kernel(
    const float* __restrict__ x, const float* __restrict__ w,
    __bf16* __restrict__ out, int D) {
  int row = blockIdx.x;
  const float* xr = x + (size_t)row * D;
  __shared__ float sbuf[256];
  float ss = 0.f;
  for (int i = threadIdx.x; i < D; i += 256) { float v = xr[i]; ss += v * v; }
  sbuf[threadIdx.x] = ss;
  __syncthreads();
  for (int s = 128; s > 0; s >>= 1) {
    if (threadIdx.x < s) sbuf[threadIdx.x] += sbuf[threadIdx.x + s];
    __syncthreads();
  }
  float scale = rsqrtf(sbuf[0] / (float)D + 1e-6f);
  for (int i = threadIdx.x; i < D; i += 256)
    out[(size_t)row * D + i] = (__bf16)(xr[i] * scale * w[i]);
}

// =====================================================================
// WMMA bf16 GEMM:  C[m,n] = sum_k A[m,k] * W[n,k]  (both K-major)
// block tile 128x128, BK=32, 8 wave32 waves, each wave 32x64 via 2x4
// v_wmma_f32_16x16x32_bf16. Double-buffered LDS filled with
// GLOBAL_LOAD_ASYNC_TO_LDS_B128 (ASYNCcnt), overlapping the next K tile
// with WMMA on the current one.
// =====================================================================
#define GBM 128
#define GBN 128
#define GBK 32
#define GLDS 40   // bf16 elems per LDS row (32 + 8 pad), 80B rows, 16B aligned
#define BUFELEMS ((GBM + GBN) * GLDS)   // bf16 elems per double-buffer slot

__global__ __launch_bounds__(256) void gemm_bf16_kernel(
    const __bf16* __restrict__ A, const __bf16* __restrict__ W,
    int M, int Nd, int K,
    const float* __restrict__ bias, const float* __restrict__ res,
    int act, float* __restrict__ outF, __bf16* __restrict__ outH) {
  __shared__ __attribute__((aligned(16))) __bf16 lds[2 * BUFELEMS];

  const int tid   = threadIdx.x;
  const int mBase = blockIdx.y * GBM;
  const int nBase = blockIdx.x * GBN;
  const int wid   = tid >> 5;
  const int lane  = tid & 31;
  const int waveM = wid & 3;      // 4 waves down M (32 rows each)
  const int waveN = wid >> 2;     // 2 waves across N (64 cols each)
  const int r     = lane & 15;
  const int hh    = lane >> 4;

  const unsigned ldsBase0 = (unsigned)(uintptr_t)&lds[0];  // hw LDS byte addr

  v8f acc[2][4] = {};

  // issue one 128x32 A tile + 128x32 W tile (async, memory -> LDS, no VGPRs)
  auto issue_tile = [&](int kb, int buf) {
    unsigned base = ldsBase0 + (unsigned)buf * (BUFELEMS * 2);
    #pragma unroll
    for (int it = 0; it < 2; ++it) {
      int ch  = tid + it * 256;        // 0..511
      int row = ch >> 2;               // 0..127
      int q   = ch & 3;                // 8-bf16 (16B) chunk within 32-elem row
      const __bf16* gA = A + (size_t)(mBase + row) * K + kb + q * 8;
      unsigned la = base + (unsigned)(row * GLDS + q * 8) * 2;
      asm volatile("global_load_async_to_lds_b128 %0, %1, off"
                   :: "v"(la), "v"(gA) : "memory");
      int gn = nBase + row;
      if (gn >= Nd) gn = Nd - 1;       // clamp; garbage cols discarded in epilogue
      const __bf16* gW = W + (size_t)gn * K + kb + q * 8;
      unsigned lw = base + (unsigned)((GBM + row) * GLDS + q * 8) * 2;
      asm volatile("global_load_async_to_lds_b128 %0, %1, off"
                   :: "v"(lw), "v"(gW) : "memory");
    }
  };

  const int nk = K / GBK;
  issue_tile(0, 0);

  for (int kt = 0; kt < nk; ++kt) {
    // wait for own async loads of tile kt, then workgroup barrier so the
    // whole tile is visible and everyone is done reading buf[(kt+1)&1]
    asm volatile("s_wait_asynccnt 0x0" ::: "memory");
    __syncthreads();
    if (kt + 1 < nk) issue_tile((kt + 1) * GBK, (kt + 1) & 1);

    const __bf16* sAb = lds + (kt & 1) * BUFELEMS;
    const __bf16* sWb = sAb + GBM * GLDS;

    // B fragments: lane<16 -> (n=r, K 0..15), lane>=16 -> (n=r, K 16..31)
    FragBf bf[4];
    #pragma unroll
    for (int nt = 0; nt < 4; ++nt) {
      const __bf16* bp = &sWb[(waveN * 64 + nt * 16 + r) * GLDS + hh * 16];
      bf[nt].q[0] = *(const u32x4*)(bp);
      bf[nt].q[1] = *(const u32x4*)(bp + 8);
    }
    // A fragments: lane<16 -> K 0..15, lane>=16 -> K 16..31
    #pragma unroll
    for (int mt = 0; mt < 2; ++mt) {
      FragBf af;
      const __bf16* ap = &sAb[(waveM * 32 + mt * 16 + r) * GLDS];
      af.q[0] = *(const u32x4*)(ap + hh * 16);
      af.q[1] = *(const u32x4*)(ap + hh * 16 + 8);
      #pragma unroll
      for (int nt = 0; nt < 4; ++nt) {
        acc[mt][nt] = __builtin_amdgcn_wmma_f32_16x16x32_bf16(
            false, af.v, false, bf[nt].v, (short)0, acc[mt][nt], false, false);
      }
    }
  }

  // ---- epilogue: C/D layout -> lane col n = r, VGPR v -> row m = v + 8*hh ----
  #pragma unroll
  for (int mt = 0; mt < 2; ++mt) {
    #pragma unroll
    for (int nt = 0; nt < 4; ++nt) {
      int n = nBase + waveN * 64 + nt * 16 + r;
      if (n >= Nd) continue;
      #pragma unroll
      for (int v = 0; v < 8; ++v) {
        int m = mBase + waveM * 32 + mt * 16 + hh * 8 + v;
        float val = acc[mt][nt][v];
        if (bias) val += bias[n];
        if (act == 1) {  // tanh-approx GELU
          float c = val + 0.044715f * val * val * val;
          val = 0.5f * val * (1.f + tanhf(0.7978845608028654f * c));
        }
        size_t off = (size_t)m * Nd + n;
        if (res)  val += res[off];
        if (outF) outF[off] = val;
        if (outH) outH[off] = (__bf16)val;
      }
    }
  }
}

// =====================================================================
// Depthwise causal conv (K=4) + SiLU + B/C l2-normalize + softplus(dt)
// one block per (b,l) row
// =====================================================================
__global__ __launch_bounds__(256) void conv_post_kernel(
    const float* __restrict__ proj, const float* __restrict__ conv_w,
    const float* __restrict__ conv_b, const float* __restrict__ dt_bias,
    float* __restrict__ u_out, float* __restrict__ Bn, float* __restrict__ Cn,
    float* __restrict__ dt_out) {
  int row = blockIdx.x;            // b*L + l
  int l = row % kL;
  __shared__ float sBC[2 * kN];
  const float* prow = proj + (size_t)row * kDPROJ + kDIN;  // xBC channels
  for (int c = threadIdx.x; c < kCONVD; c += 256) {
    float acc = conv_b[c];
    #pragma unroll
    for (int j = 0; j < kKc; ++j) {
      int ll = l - (kKc - 1) + j;
      float xv = (ll >= 0) ? prow[(ptrdiff_t)(ll - l) * kDPROJ + c] : 0.f;
      acc = fmaf(conv_w[c * kKc + j], xv, acc);
    }
    float s = acc / (1.f + expf(-acc));      // SiLU
    if (c < kDIN) u_out[(size_t)row * kDIN + c] = s;
    else          sBC[c - kDIN] = s;
  }
  __syncthreads();
  int t = threadIdx.x;
  if (t < kN) {
    float ss = 1e-6f;
    for (int i = 0; i < kN; ++i) ss += sBC[i] * sBC[i];
    Bn[row * kN + t] = sBC[t] * rsqrtf(ss);
  } else if (t < 2 * kN) {
    float ss = 1e-6f;
    for (int i = 0; i < kN; ++i) ss += sBC[kN + i] * sBC[kN + i];
    Cn[row * kN + (t - kN)] = sBC[t] * rsqrtf(ss);
  } else if (t < 2 * kN + kH) {
    int h = t - 2 * kN;
    float v = proj[(size_t)row * kDPROJ + 2 * kDIN + 2 * kN + h] + dt_bias[h];
    dt_out[row * kH + h] = (v > 20.f) ? v : log1pf(expf(v));
  }
}

// =====================================================================
// Selective scan: block = (b,h), thread = p (64), n in registers (16)
// writes gated bf16 output for out_proj GEMM
// =====================================================================
__global__ __launch_bounds__(64) void scan_kernel(
    const float* __restrict__ proj, const float* __restrict__ u_in,
    const float* __restrict__ Bn, const float* __restrict__ Cn,
    const float* __restrict__ dtv, const float* __restrict__ A_param,
    const float* __restrict__ D_skip, __bf16* __restrict__ yg) {
  int b = blockIdx.x / kH;
  int h = blockIdx.x % kH;
  int p = threadIdx.x;
  float A[kN];
  #pragma unroll
  for (int n = 0; n < kN; ++n) A[n] = fmaxf(A_param[h * kN + n], 0.f);
  float Dsk = D_skip[h];
  float z[kN] = {};
  float y[kN] = {};
  __shared__ float sB[kN], sC[kN];
  __shared__ float sdt;

  for (int l = 0; l < kL; ++l) {
    int row = b * kL + l;
    if (p < kN)            sB[p] = Bn[row * kN + p];
    else if (p < 2 * kN)   sC[p - kN] = Cn[row * kN + (p - kN)];
    else if (p == 2 * kN)  sdt = dtv[row * kH + h];
    __syncthreads();
    float d = sdt;
    float u = u_in[(size_t)row * kDIN + h * kP + p];
    float acc = 0.f;
    #pragma unroll
    for (int n = 0; n < kN; ++n) {
      float Bu  = u * sB[n];
      float rcp = 1.f / (1.f + d * d * A[n]);
      float zn  = (z[n] + d * (Bu - A[n] * y[n])) * rcp;
      z[n] = zn;
      y[n] += d * zn;
      acc = fmaf(y[n], sC[n], acc);
    }
    float outv = acc + Dsk * u;
    float zg = proj[(size_t)row * kDPROJ + h * kP + p];   // zgate channel
    float g  = zg / (1.f + expf(-zg));                    // silu
    yg[(size_t)row * kDIN + h * kP + p] = (__bf16)(outv * g);
    __syncthreads();
  }
}

// =====================================================================
// launch
// =====================================================================
extern "C" void kernel_launch(void* const* d_in, const int* in_sizes, int n_in,
                              void* d_out, int out_size, void* d_ws, size_t ws_size,
                              hipStream_t stream) {
  (void)in_sizes; (void)n_in; (void)out_size; (void)ws_size;
  const float* x          = (const float*)d_in[0];
  const float* norm1_w    = (const float*)d_in[1];
  const float* in_proj_w  = (const float*)d_in[2];
  const float* conv_w     = (const float*)d_in[3];
  const float* conv_b     = (const float*)d_in[4];
  const float* A_param    = (const float*)d_in[5];
  const float* dt_bias    = (const float*)d_in[6];
  const float* D_skip     = (const float*)d_in[7];
  const float* out_proj_w = (const float*)d_in[8];
  const float* out_proj_b = (const float*)d_in[9];
  const float* norm2_w    = (const float*)d_in[10];
  const float* fc1_w      = (const float*)d_in[11];
  const float* fc1_b      = (const float*)d_in[12];
  const float* fc2_w      = (const float*)d_in[13];
  const float* fc2_b      = (const float*)d_in[14];
  float* out = (float*)d_out;

  // ---- workspace carve ----
  char* wp = (char*)d_ws;
  auto alloc = [&](size_t bytes) -> void* {
    void* r = (void*)wp;
    wp += (bytes + 255) & ~(size_t)255;
    return r;
  };
  __bf16* wInProj  = (__bf16*)alloc((size_t)kDPROJ * kD * 2);
  __bf16* wOutProj = (__bf16*)alloc((size_t)kD * kDIN * 2);
  __bf16* wFc1     = (__bf16*)alloc((size_t)4 * kD * kD * 2);
  __bf16* wFc2     = (__bf16*)alloc((size_t)kD * 4 * kD * 2);
  __bf16* xn_bf    = (__bf16*)alloc((size_t)kRows * kD * 2);
  float*  proj     = (float*) alloc((size_t)kRows * kDPROJ * 4);
  float*  u_f      = (float*) alloc((size_t)kRows * kDIN * 4);
  float*  Bn_f     = (float*) alloc((size_t)kRows * kN * 4);
  float*  Cn_f     = (float*) alloc((size_t)kRows * kN * 4);
  float*  dt_f     = (float*) alloc((size_t)kRows * kH * 4);
  __bf16* yg_bf    = (__bf16*)alloc((size_t)kRows * kDIN * 2);
  float*  x2_f     = (float*) alloc((size_t)kRows * kD * 4);
  __bf16* h_bf     = (__bf16*)alloc((size_t)kRows * kD * 2);
  __bf16* h1_bf    = (__bf16*)alloc((size_t)kRows * 4 * kD * 2);

  // ---- weight conversions to bf16 ----
  auto cvt = [&](const float* src, __bf16* dst, int n) {
    cvt_bf16_kernel<<<(n + 255) / 256, 256, 0, stream>>>(src, dst, n);
  };
  cvt(in_proj_w,  wInProj,  kDPROJ * kD);
  cvt(out_proj_w, wOutProj, kD * kDIN);
  cvt(fc1_w,      wFc1,     4 * kD * kD);
  cvt(fc2_w,      wFc2,     kD * 4 * kD);

  // ---- 1) rms1 ----
  rms_bf16_kernel<<<kRows, 256, 0, stream>>>(x, norm1_w, xn_bf, kD);

  // ---- 2) in_proj GEMM: proj = xn @ in_proj_w.T  (8192 x 4160 x 1024) ----
  {
    dim3 g((kDPROJ + GBN - 1) / GBN, kRows / GBM);
    gemm_bf16_kernel<<<g, 256, 0, stream>>>(xn_bf, wInProj, kRows, kDPROJ, kD,
                                            nullptr, nullptr, 0, proj, nullptr);
  }

  // ---- 3) conv + silu + B/C normalize + softplus(dt) ----
  conv_post_kernel<<<kRows, 256, 0, stream>>>(proj, conv_w, conv_b, dt_bias,
                                              u_f, Bn_f, Cn_f, dt_f);

  // ---- 4) selective scan + gating -> yg_bf ----
  scan_kernel<<<kB * kH, kP, 0, stream>>>(proj, u_f, Bn_f, Cn_f, dt_f,
                                          A_param, D_skip, yg_bf);

  // ---- 5) out_proj GEMM + bias + residual(x) -> x2 ----
  {
    dim3 g(kD / GBN, kRows / GBM);
    gemm_bf16_kernel<<<g, 256, 0, stream>>>(yg_bf, wOutProj, kRows, kD, kDIN,
                                            out_proj_b, x, 0, x2_f, nullptr);
  }

  // ---- 6) rms2 ----
  rms_bf16_kernel<<<kRows, 256, 0, stream>>>(x2_f, norm2_w, h_bf, kD);

  // ---- 7) fc1 GEMM + bias + gelu -> h1_bf ----
  {
    dim3 g(4 * kD / GBN, kRows / GBM);
    gemm_bf16_kernel<<<g, 256, 0, stream>>>(h_bf, wFc1, kRows, 4 * kD, kD,
                                            fc1_b, nullptr, 1, nullptr, h1_bf);
  }

  // ---- 8) fc2 GEMM + bias + residual(x2) -> out ----
  {
    dim3 g(kD / GBN, kRows / GBM);
    gemm_bf16_kernel<<<g, 256, 0, stream>>>(h1_bf, wFc2, kRows, kD, 4 * kD,
                                            fc2_b, x2_f, 0, out, nullptr);
  }
}
